// LSTMAttn_19284403159558
// MI455X (gfx1250) — matrix operand order repaired and checked by reading, not compiled, not run on hardware
//
#include <hip/hip_runtime.h>
#include <stdint.h>

#define H       512
#define BATCH   128
#define T_ENC   512
#define PRED    32
#define NWG     16
#define SLICE   32            // hidden units per workgroup
#define GR      128           // gate rows per workgroup (4 gates * SLICE)
#define THREADS 256           // 8 wave32s

typedef __attribute__((ext_vector_type(16))) __bf16 v16bf;
typedef __attribute__((ext_vector_type(8)))  float  v8f;

union Frag { uint32_t w[8]; v16bf v; };

__device__ __forceinline__ uint16_t f2bf(float f) {
  union { float f; uint32_t u; } v; v.f = f;
  uint32_t r = v.u + 0x7FFFu + ((v.u >> 16) & 1u);   // round-to-nearest-even
  return (uint16_t)(r >> 16);
}
__device__ __forceinline__ float bf2f(uint16_t h) {
  union { uint32_t u; float f; } v; v.u = ((uint32_t)h) << 16; return v.f;
}
__device__ __forceinline__ float sigmoidf(float x) {
  return 1.0f / (1.0f + __expf(-x));
}

// ---------------------------------------------------------------------------
// Init: zero the grid-barrier counter, seed h_buf[0] = bf16(h0)
// ---------------------------------------------------------------------------
__global__ void lstm_init_kernel(const float* __restrict__ h0,
                                 uint16_t* __restrict__ hbuf0,
                                 unsigned* __restrict__ bar) {
  int i = blockIdx.x * blockDim.x + threadIdx.x;
  if (i == 0) *bar = 0u;
  for (int e = i; e < BATCH * H; e += gridDim.x * blockDim.x)
    hbuf0[e] = f2bf(h0[e]);
}

// ---------------------------------------------------------------------------
// Persistent LSTM encoder. 16 workgroups, each owns hidden slice [j0, j0+32).
// Whh slice (128 rows x 512 cols) lives in LDS as bf16 for the whole run.
// Per step: gates = h @ Whh_slice^T via v_wmma_f32_16x16x32_bf16,
// then elementwise LSTM cell update, then a grid-wide atomic barrier.
// h is double-buffered bf16 in global workspace.
// ---------------------------------------------------------------------------
__global__ __launch_bounds__(THREADS, 1)
void lstm_encoder_kernel(const float* __restrict__ x_seq,
                         const float* __restrict__ c0,
                         const float* __restrict__ Wih,
                         const float* __restrict__ Whh,
                         const float* __restrict__ bih,
                         const float* __restrict__ bhh,
                         uint16_t* __restrict__ h_buf,   // [2][BATCH*H] bf16
                         float*    __restrict__ h_enc,   // [BATCH*H] f32
                         uint16_t* __restrict__ enc,     // [T_ENC*BATCH*H] bf16
                         unsigned* __restrict__ bar) {
  extern __shared__ char smem[];
  uint16_t* wgt  = (uint16_t*)smem;                                  // 128 KB
  float* gates   = (float*)(smem + GR * H * 2);                      //  64 KB
  float* c_s     = (float*)(smem + GR * H * 2 + BATCH * GR * 4);     //  16 KB
  float* bias_s  = (float*)(smem + GR * H * 2 + BATCH * GR * 4 + BATCH * SLICE * 4);
  float* wih_s   = bias_s + GR;

  const int tid = threadIdx.x;
  const int g   = blockIdx.x;
  const int j0  = g * SLICE;

  // --- stage weight slice (rows: i[0:32) f[32:64) g[64:96) o[96:128)) ---
  for (int e = tid; e < GR * H; e += THREADS) {
    int n = e >> 9, k = e & (H - 1);
    int grow = (n >> 5) * H + j0 + (n & (SLICE - 1));   // global Whh row
    wgt[n * H + k] = f2bf(Whh[grow * H + k]);
  }
  for (int n = tid; n < GR; n += THREADS) {
    int grow = (n >> 5) * H + j0 + (n & (SLICE - 1));
    bias_s[n] = bih[grow] + bhh[grow];
    wih_s[n]  = Wih[grow];                               // I == 1
  }
  for (int e = tid; e < BATCH * SLICE; e += THREADS) {
    int b = e >> 5, s = e & (SLICE - 1);
    c_s[e] = c0[b * H + j0 + s];
  }
  __syncthreads();

  const int wv   = tid >> 5;        // wave id 0..7 -> N-tile
  const int lane = tid & 31;
  const int half = lane >> 4;       // lane half per WMMA layout
  const int mrow = lane & 15;
  const uint32_t* wgt32 = (const uint32_t*)wgt;
  // B fragment: lane = column n (= gate row), K 0..15 / 16..31 by lane half
  const uint32_t* brow_base = wgt32 + (wv * 16 + mrow) * (H / 2) + half * 8;

  for (int t = 0; t < T_ENC; ++t) {
    const uint32_t* hread  = (const uint32_t*)(h_buf + (size_t)(t & 1) * (BATCH * H));
    uint16_t*       hwrite = h_buf + (size_t)((t + 1) & 1) * (BATCH * H);

    v8f acc[8] = {};

    for (int kb = 0; kb < 16; ++kb) {           // K = 512 in chunks of 32
      const int k32 = kb * 16;                  // dword offset of k0
      Frag bf;
      const uint32_t* bp = brow_base + k32;
      #pragma unroll
      for (int i = 0; i < 8; ++i) bf.w[i] = bp[i];
      #pragma unroll
      for (int mt = 0; mt < 8; ++mt) {          // M = 128 batch rows
        Frag af;
        // A fragment: lane = row M, dwords [k0/2+8h .. +7]
        const uint32_t* ap = hread + (mt * 16 + mrow) * (H / 2) + k32 + half * 8;
        #pragma unroll
        for (int i = 0; i < 4; ++i) af.w[i]     = ap[i];
        #pragma unroll
        for (int i = 0; i < 4; ++i) af.w[4 + i] = ap[4 + i];
        acc[mt] = __builtin_amdgcn_wmma_f32_16x16x32_bf16(
            false, af.v, false, bf.v, (short)0, acc[mt], false, false);
      }
    }

    // C layout: VGPR r -> M = r + 8*half, N = mrow
    #pragma unroll
    for (int mt = 0; mt < 8; ++mt)
      #pragma unroll
      for (int r = 0; r < 8; ++r)
        gates[(mt * 16 + r + half * 8) * GR + wv * 16 + mrow] = acc[mt][r];
    __syncthreads();

    // --- LSTM cell update: 4096 (b,s) pairs, 16 per thread ---
    for (int i = 0; i < 16; ++i) {
      int idx = tid + (i << 8);
      int b = idx >> 5, s = idx & (SLICE - 1);
      float x = x_seq[t * BATCH + b];
      const float* grow = gates + b * GR;
      float iv = grow[s]             + x * wih_s[s]             + bias_s[s];
      float fv = grow[SLICE + s]     + x * wih_s[SLICE + s]     + bias_s[SLICE + s];
      float gv = grow[2 * SLICE + s] + x * wih_s[2 * SLICE + s] + bias_s[2 * SLICE + s];
      float ov = grow[3 * SLICE + s] + x * wih_s[3 * SLICE + s] + bias_s[3 * SLICE + s];
      float c = sigmoidf(fv) * c_s[idx] + sigmoidf(iv) * tanhf(gv);
      float h = sigmoidf(ov) * tanhf(c);
      c_s[idx] = c;
      uint16_t hb = f2bf(h);
      hwrite[b * H + j0 + s] = hb;
      enc[((size_t)t * BATCH + b) * H + j0 + s] = hb;
      if (t == T_ENC - 1) h_enc[b * H + j0 + s] = h;
    }

    // --- grid-wide barrier (monotonic counter, s_sleep backoff) ---
    __syncthreads();
    if (tid == 0) {
      __threadfence();
      __hip_atomic_fetch_add(bar, 1u, __ATOMIC_ACQ_REL, __HIP_MEMORY_SCOPE_AGENT);
      unsigned target = (unsigned)(NWG * (t + 1));
      while (__hip_atomic_load(bar, __ATOMIC_ACQUIRE, __HIP_MEMORY_SCOPE_AGENT) < target)
        __builtin_amdgcn_s_sleep(1);
    }
    __syncthreads();
  }
}

// ---------------------------------------------------------------------------
// Decoder: one workgroup per batch element (fully parallel across batch).
// 32 serial steps of RNN cell + dot-product attention + linear/ReLU.
// enc is bf16 and fits in L2 (64 MB), so attention streams from L2.
// ---------------------------------------------------------------------------
__global__ __launch_bounds__(THREADS, 1)
void attn_decoder_kernel(const float* __restrict__ x_seq,
                         const float* __restrict__ rWih,
                         const float* __restrict__ rWhh,
                         const float* __restrict__ rbih,
                         const float* __restrict__ rbhh,
                         const float* __restrict__ linW,
                         const float* __restrict__ linb,
                         const float* __restrict__ h_enc,
                         const uint16_t* __restrict__ enc,
                         float* __restrict__ out) {
  __shared__ float hA[H], hB[H], wsc[T_ENC], ctx[H], red[THREADS];
  __shared__ float xsh;
  const int tid = threadIdx.x;
  const int b = blockIdx.x;
  float* hcur = hA; float* hnxt = hB;

  for (int j = tid; j < H; j += THREADS) hA[j] = h_enc[b * H + j];
  if (tid == 0) xsh = x_seq[(size_t)T_ENC * BATCH + b];   // input_seq[-1]
  __syncthreads();

  for (int step = 0; step < PRED; ++step) {
    float x = xsh;
    // RNN cell
    for (int j = tid; j < H; j += THREADS) {
      float a = x * rWih[j] + rbih[j] + rbhh[j];
      const float* wr = rWhh + (size_t)j * H;
      #pragma unroll 8
      for (int k = 0; k < H; ++k) a += hcur[k] * wr[k];
      hnxt[j] = tanhf(a);
    }
    __syncthreads();
    // attention scores
    for (int t2 = tid; t2 < T_ENC; t2 += THREADS) {
      const uint16_t* er = enc + ((size_t)t2 * BATCH + b) * H;
      float s = 0.f;
      #pragma unroll 8
      for (int k = 0; k < H; ++k) s += bf2f(er[k]) * hnxt[k];
      wsc[t2] = s;
    }
    __syncthreads();
    // softmax over 512 scores
    float m = fmaxf(wsc[tid], wsc[tid + THREADS]);
    red[tid] = m; __syncthreads();
    for (int off = THREADS / 2; off > 0; off >>= 1) {
      if (tid < off) red[tid] = fmaxf(red[tid], red[tid + off]);
      __syncthreads();
    }
    m = red[0]; __syncthreads();
    float e0 = __expf(wsc[tid] - m), e1 = __expf(wsc[tid + THREADS] - m);
    red[tid] = e0 + e1; __syncthreads();
    for (int off = THREADS / 2; off > 0; off >>= 1) {
      if (tid < off) red[tid] += red[tid + off];
      __syncthreads();
    }
    float inv = 1.f / red[0]; __syncthreads();
    wsc[tid] = e0 * inv; wsc[tid + THREADS] = e1 * inv;
    __syncthreads();
    // context vector
    for (int hh = tid; hh < H; hh += THREADS) {
      float a = 0.f;
      #pragma unroll 4
      for (int t2 = 0; t2 < T_ENC; ++t2)
        a += wsc[t2] * bf2f(enc[((size_t)t2 * BATCH + b) * H + hh]);
      ctx[hh] = a;
    }
    __syncthreads();
    // linear + ReLU (O == 1)
    red[tid] = ctx[tid] * linW[tid] + ctx[tid + THREADS] * linW[tid + THREADS];
    __syncthreads();
    for (int off = THREADS / 2; off > 0; off >>= 1) {
      if (tid < off) red[tid] += red[tid + off];
      __syncthreads();
    }
    if (tid == 0) {
      float o = red[0] + linb[0];
      o = o > 0.f ? o : 0.f;
      out[step * BATCH + b] = o;
      xsh = o;
    }
    __syncthreads();
    float* tmp = hcur; hcur = hnxt; hnxt = tmp;
  }
}

// ---------------------------------------------------------------------------
// Launch. Workspace layout (needs ~68 MB):
//   [0,256)                        : grid barrier counter
//   [256, +256KB)                  : h double buffer, bf16 [2][128*512]
//   [+256KB, +256KB)               : final encoder h, f32 [128*512]
//   [..., +64MB)                   : encoder outputs, bf16 [512*128*512]
// ---------------------------------------------------------------------------
extern "C" void kernel_launch(void* const* d_in, const int* in_sizes, int n_in,
                              void* d_out, int out_size, void* d_ws, size_t ws_size,
                              hipStream_t stream) {
  const float* x_seq = (const float*)d_in[0];
  const float* h0    = (const float*)d_in[1];
  const float* c0    = (const float*)d_in[2];
  const float* lWih  = (const float*)d_in[3];
  const float* lWhh  = (const float*)d_in[4];
  const float* lbih  = (const float*)d_in[5];
  const float* lbhh  = (const float*)d_in[6];
  const float* rWih  = (const float*)d_in[7];
  const float* rWhh  = (const float*)d_in[8];
  const float* rbih  = (const float*)d_in[9];
  const float* rbhh  = (const float*)d_in[10];
  const float* linW  = (const float*)d_in[11];
  const float* linb  = (const float*)d_in[12];
  (void)in_sizes; (void)n_in; (void)out_size; (void)ws_size;

  char* ws = (char*)d_ws;
  unsigned* bar   = (unsigned*)ws;
  uint16_t* h_buf = (uint16_t*)(ws + 256);
  float*    h_enc = (float*)(ws + 256 + (size_t)2 * BATCH * H * 2);
  uint16_t* enc   = (uint16_t*)(ws + 256 + (size_t)2 * BATCH * H * 2 + (size_t)BATCH * H * 4);

  const size_t SMEM = (size_t)GR * H * 2 + (size_t)BATCH * GR * 4 +
                      (size_t)BATCH * SLICE * 4 + (size_t)GR * 4 * 2;  // 214016 B
  (void)hipFuncSetAttribute((const void*)lstm_encoder_kernel,
                            hipFuncAttributeMaxDynamicSharedMemorySize, (int)SMEM);

  lstm_init_kernel<<<64, THREADS, 0, stream>>>(h0, h_buf, bar);
  lstm_encoder_kernel<<<NWG, THREADS, SMEM, stream>>>(
      x_seq, c0, lWih, lWhh, lbih, lbhh, h_buf, h_enc, enc, bar);
  attn_decoder_kernel<<<BATCH, THREADS, 0, stream>>>(
      x_seq, rWih, rWhh, rbih, rbhh, linW, linb, h_enc, enc, (float*)d_out);
}
